// LSTMTagger_48902497632724
// MI455X (gfx1250) — matrix operand order repaired and checked by reading, not compiled, not run on hardware
//
#include <hip/hip_runtime.h>
#include <hip/hip_bf16.h>
#include <math.h>

// ---------------------------------------------------------------------------
// Problem constants (from reference): B=64, T=256, D=256, H=512, 4H=2048
// ---------------------------------------------------------------------------
#define BB   64
#define TT   256
#define DD   256
#define HH   512
#define G4   2048   // 4*H

typedef __attribute__((ext_vector_type(16))) __bf16 v16bf;
typedef __attribute__((ext_vector_type(8)))  float  v8f;
typedef __attribute__((ext_vector_type(4)))  unsigned int u32x4;
typedef __attribute__((ext_vector_type(8)))  int i32x8;
typedef __attribute__((ext_vector_type(4)))  int i32x4;

#define AS1 __attribute__((address_space(1)))
#define AS3 __attribute__((address_space(3)))

// Pointer-to-AS1/AS3 vector types for the async LDS-DMA builtin
typedef AS1 i32x4* gv4p;   // global (device) memory, v4i32 pointee
typedef AS3 i32x4* lv4p;   // LDS, v4i32 pointee

struct alignas(16) Q4 { unsigned int u[4]; };   // trivial 16-byte chunk
union Frag { Q4 q[2]; v16bf v; };

// Feature guards (also keep host-pass compilation on the fallback path)
#ifndef __has_builtin
#define __has_builtin(x) 0
#endif
#if __has_builtin(__builtin_amdgcn_tensor_load_to_lds) && \
    __has_builtin(__builtin_amdgcn_s_wait_tensorcnt)
#define HAVE_TDM 1
#else
#define HAVE_TDM 0
#endif
#if __has_builtin(__builtin_amdgcn_global_load_async_to_lds_b128)
#define HAVE_ASYNC 1
#else
#define HAVE_ASYNC 0
#endif

#if HAVE_TDM
// Tensor Data Mover: descriptor groups per CDNA5 ISA.
// ROCm 7.2 clang-22 builtin is 5-arg; therock-10.0 headers ship the 6-arg
// form (detected via the gfx1250 TDM header).
__device__ __forceinline__ void tdm_load_2d(u32x4 g0, i32x8 g1) {
    i32x4 z4 = {0, 0, 0, 0};
#if __has_include(<hip/amd_detail/amd_gfx1250_TDM.h>)
    i32x8 z8 = {0, 0, 0, 0, 0, 0, 0, 0};
    __builtin_amdgcn_tensor_load_to_lds(g0, g1, z4, z4, z8, 0);
#else
    __builtin_amdgcn_tensor_load_to_lds(g0, g1, z4, z4, 0);
#endif
}
#endif

#if HAVE_ASYNC
__device__ __forceinline__ void wait_async0() {
#if __has_builtin(__builtin_amdgcn_s_wait_asynccnt)
    __builtin_amdgcn_s_wait_asynccnt(0);
#else
    asm volatile("s_wait_asynccnt 0" ::: "memory");
#endif
}
#endif

// ---------------------------------------------------------------------------
// Kernel 1: fp32 -> bf16 conversion of x and W_ih, + barrier init
// ---------------------------------------------------------------------------
__global__ void convert_init_kernel(const float* __restrict__ x,
                                    const float* __restrict__ wih,
                                    __bf16* __restrict__ xb,
                                    __bf16* __restrict__ wib,
                                    int* __restrict__ bar) {
    long long idx = (long long)blockIdx.x * blockDim.x + threadIdx.x;
    const long long nx = (long long)BB * TT * DD;       // 4,194,304
    const long long nw = (long long)G4 * DD;            //   524,288
    if (idx == 0) { bar[0] = 0; bar[1] = 0; }           // grid-barrier state
    if (idx < nx) {
        xb[idx] = (__bf16)x[idx];
    } else if (idx < nx + nw) {
        long long j = idx - nx;
        wib[j] = (__bf16)wih[j];
    }
}

// ---------------------------------------------------------------------------
// Kernel 2: XW[m,n] = x[m,:] @ W_ih[n,:]^T + b_ih[n] + b_hh[n]
//   M = B*T = 16384, N = 4H = 2048, K = D = 256
//   Block: 256 threads (8 waves). Block tile 128(M) x 128(N).
//   The block's whole 128x256 bf16 W_ih panel (one TDM DMA, padded rows to
//   dodge LDS bank conflicts) is resident in LDS; the K-loop is barrier-free:
//   per wave 1 A-frag global load + 8 LDS B-frags + 8 WMMAs per K-step.
// ---------------------------------------------------------------------------
__global__ void xw_gemm_kernel(const __bf16* __restrict__ xb,
                               const __bf16* __restrict__ wib,
                               const float*  __restrict__ b_ih,
                               const float*  __restrict__ b_hh,
                               float* __restrict__ XW) {
    constexpr int BROW = 264;                 // 256 + 8 bf16 pad (4 dwords/row)
    __shared__ __bf16 ldsb[128 * BROW];       // ~66 KB (CDNA5: 320 KB/WGP)
    const int tid  = threadIdx.x;
    const int wave = tid >> 5;                // wave32
    const int lane = tid & 31;
    const int col  = lane & 15;
    const int hi   = lane >> 4;               // 0: lanes 0-15, 1: 16-31
    const int mbase = blockIdx.x * 128;
    const int nbase = blockIdx.y * 128;
    const int m = mbase + wave * 16 + col;    // A-row for this lane

#if HAVE_TDM
    if (tid == 0) {
        // D# group 0: count=1 | lds_addr | 57b global_addr | type=2
        unsigned lds_off = (unsigned)(unsigned long long)(AS3 __bf16*)&ldsb[0];
        unsigned long long ga =
            (unsigned long long)(uintptr_t)(wib + (size_t)nbase * DD);
        u32x4 g0 = { 1u, lds_off,
                     (unsigned)(ga & 0xffffffffu),
                     (unsigned)(((ga >> 32) & 0x01ffffffu) | 0x80000000u) };
        // D# group 1: data_size=2B(code1), pad_enable, pad_interval=6 (128 dw),
        // pad_amount=3 (4 dw); tensor 256 x 2048 (stride 256), tile 256 x 128.
        i32x8 g1 = { (int)0x07910000,
                     (int)(256u  << 16),      // tensor_dim0 lo16 << 16
                     (int)(2048u << 16),      // tensor_dim1 lo16 << 16
                     (int)(256u  << 16),      // tile_dim0 << 16
                     128,                     // tile_dim1
                     256,                     // tensor_dim0_stride lo32
                     0, 0 };
        tdm_load_2d(g0, g1);
    }
    __builtin_amdgcn_s_wait_tensorcnt(0);
#else
    // Fallback: manual panel copy, 32 B per thread-chunk, padded layout
    for (int e = tid; e < 128 * 16; e += 256) {
        int r = e >> 4, c = (e & 15) * 16;
        const Q4* src = reinterpret_cast<const Q4*>(wib + (size_t)(nbase + r) * DD + c);
        Q4* dst = reinterpret_cast<Q4*>(&ldsb[r * BROW + c]);
        dst[0] = src[0];
        dst[1] = src[1];
    }
#endif
    __syncthreads();

    v8f acc[8] = {};
    for (int kb = 0; kb < DD; kb += 32) {
        if (kb + 32 < DD) {
            __builtin_prefetch(xb + (size_t)m * DD + kb + 32, 0, 1);  // global_prefetch_b8
        }
        // A fragment (16x32 bf16): low lanes K {0..15}, high {16..31}
        Frag a;
        const int koff = hi ? 16 : 0;
        a.q[0] = reinterpret_cast<const Q4*>(xb + (size_t)m * DD + kb + koff)[0];
        a.q[1] = reinterpret_cast<const Q4*>(xb + (size_t)m * DD + kb + 8 + koff)[0];
#pragma unroll
        for (int nt = 0; nt < 8; ++nt) {
            // B fragment (32x16): lane holds column n = col, K half per hi
            Frag bf;
            const Q4* bs = reinterpret_cast<const Q4*>(
                &ldsb[(nt * 16 + col) * BROW + kb + hi * 16]);
            bf.q[0] = bs[0];
            bf.q[1] = bs[1];
            acc[nt] = __builtin_amdgcn_wmma_f32_16x16x32_bf16(
                false, a.v, false, bf.v, (short)0, acc[nt], false, false);
        }
    }

    // Epilogue: C layout — lane col = N, VGPR r = M row (+8 for high lanes)
#pragma unroll
    for (int nt = 0; nt < 8; ++nt) {
        const int n = nbase + nt * 16 + col;
        const float bias = b_ih[n] + b_hh[n];
        const int mrow = mbase + wave * 16 + hi * 8;
#pragma unroll
        for (int r = 0; r < 8; ++r) {
            XW[(size_t)(mrow + r) * G4 + n] = acc[nt][r] + bias;
        }
    }
}

// ---------------------------------------------------------------------------
// Software grid barrier (16 blocks). bar[0]=arrival count, bar[1]=generation.
// ---------------------------------------------------------------------------
__device__ __forceinline__ void grid_sync(int* bar, int nblk) {
    __syncthreads();
    if (threadIdx.x == 0) {
        __threadfence();
        volatile int* gen = bar + 1;
        int g = *gen;
        if (atomicAdd(bar, 1) == nblk - 1) {
            bar[0] = 0;
            __threadfence();
            atomicExch((int*)gen, g + 1);
        } else {
            while (*gen == g) { __builtin_amdgcn_s_sleep(2); }
        }
        __threadfence();
    }
    __syncthreads();
}

__device__ __forceinline__ float sigm(float v) { return 1.0f / (1.0f + __expf(-v)); }

// ---------------------------------------------------------------------------
// Kernel 3: sequential LSTM over active timesteps.
//   16 workgroups x 256 threads. WG w owns h-elements j in [w*32, w*32+32)
//   and gate rows {g*512 + w*32 + j}. Its 128x512 fp32 W_hh slice lives in
//   LDS (256 KB, transposed [k][r] => conflict-free) — CDNA5's 320 KB LDS.
//   Per step: async-DMA h (512 floats) into LDS, 2-way-K-split dot + LDS
//   reduction, LSTM pointwise update on 32 lanes, publish 32 h values,
//   grid barrier. State (h via global, c in LDS) persists across samples.
// ---------------------------------------------------------------------------
__global__ void lstm_seq_kernel(const float* __restrict__ XW,
                                const float* __restrict__ whh,
                                const int*   __restrict__ starts,
                                const int*   __restrict__ ends,
                                float* __restrict__ hbuf,
                                int*   __restrict__ bar,
                                float* __restrict__ h_all) {
    extern __shared__ float smem[];
    float* wsl  = smem;                 // [512][128] transposed slice, 256 KB
    float* hsh  = wsl + 512 * 128;      // [512] staged h
    float* psum = hsh + 512;            // [256] partial dots
    float* gf   = psum + 256;           // [128] full gate values
    float* csh  = gf + 128;             // [32]  cell state (persists)

    const int wg  = blockIdx.x;         // 0..15
    const int tid = threadIdx.x;        // 0..255

    // Load this WG's W_hh slice, transposed: wsl[k*128 + r] = W_hh[n(r), k]
    for (int e = tid; e < 128 * 512; e += 256) {
        int r = e >> 9;                 // 0..127
        int k = e & 511;
        int gate = r >> 5, jl = r & 31;
        int n = gate * HH + wg * 32 + jl;
        wsl[k * 128 + r] = whh[(size_t)n * HH + k];
    }
    if (tid < 32) {
        csh[tid] = 0.0f;
        hbuf[wg * 32 + tid] = 0.0f;
    }
    __threadfence();
    __syncthreads();
    grid_sync(bar, 16);

    const int r    = tid & 127;
    const int half = tid >> 7;          // K half: 0 -> k 0..255, 1 -> 256..511
    const int gate = r >> 5, jl = r & 31;
    const int nrow = gate * HH + wg * 32 + jl;

    for (int b = 0; b < BB; ++b) {
        const int s = starts[b];
        const int e = ends[b];
        for (int t = s + 1; t <= e; ++t) {
            // stage full h (2 KB) into LDS
#if HAVE_ASYNC
            if (tid < 128) {
                __builtin_amdgcn_global_load_async_to_lds_b128(
                    (gv4p)(hbuf + tid * 4),
                    (lv4p)(hsh + tid * 4), 0, 0);
            }
            wait_async0();
#else
            for (int k = tid; k < HH; k += 256) hsh[k] = hbuf[k];
#endif
            __syncthreads();

            float acc = 0.0f;
            const int k0 = half * 256;
#pragma unroll 8
            for (int k = 0; k < 256; ++k) {
                acc = fmaf(wsl[(k0 + k) * 128 + r], hsh[k0 + k], acc);
            }
            psum[tid] = acc;
            __syncthreads();

            if (tid < 128) {
                gf[tid] = psum[tid] + psum[tid + 128] +
                          XW[((size_t)(b * TT + t)) * G4 + nrow];
            }
            __syncthreads();

            if (tid < 32) {
                float gi  = gf[tid];
                float gfo = gf[32 + tid];
                float gg  = gf[64 + tid];
                float go  = gf[96 + tid];
                float c = csh[tid];
                float cn = sigm(gfo) * c + sigm(gi) * tanhf(gg);
                float hn = sigm(go) * tanhf(cn);
                csh[tid] = cn;
                hbuf[wg * 32 + tid] = hn;
            }
            __threadfence();
            grid_sync(bar, 16);
        }
        // h after this sample's active steps (state carries into next sample)
        if (tid < 32) h_all[(size_t)b * HH + wg * 32 + tid] = hbuf[wg * 32 + tid];
        __syncthreads();
    }
}

// ---------------------------------------------------------------------------
// MLP head (tiny): z1 = relu(h W1^T + b1); z2 = relu(z1 W2^T + b2);
// out = sigmoid(z2 W3^T + b3); plus true_vals passthrough.
// ---------------------------------------------------------------------------
__global__ void mlp1_kernel(const float* __restrict__ h_all,
                            const float* __restrict__ W1,
                            const float* __restrict__ b1,
                            float* __restrict__ z1) {
    int idx = blockIdx.x * blockDim.x + threadIdx.x;    // 64*1024
    if (idx >= BB * 2 * HH) return;
    int b = idx / (2 * HH);
    int j = idx % (2 * HH);
    float acc = b1[j];
    const float* hv = h_all + (size_t)b * HH;
    const float* wv = W1 + (size_t)j * HH;
    for (int k = 0; k < HH; ++k) acc = fmaf(hv[k], wv[k], acc);
    z1[idx] = fmaxf(acc, 0.0f);
}

__global__ void mlp2_kernel(const float* __restrict__ z1,
                            const float* __restrict__ W2,
                            const float* __restrict__ b2,
                            float* __restrict__ z2) {
    int idx = blockIdx.x * blockDim.x + threadIdx.x;    // 64*1024
    if (idx >= BB * 2 * HH) return;
    int b = idx / (2 * HH);
    int j = idx % (2 * HH);
    float acc = b2[j];
    const float* zv = z1 + (size_t)b * (2 * HH);
    const float* wv = W2 + (size_t)j * (2 * HH);
    for (int k = 0; k < 2 * HH; ++k) acc = fmaf(zv[k], wv[k], acc);
    z2[idx] = fmaxf(acc, 0.0f);
}

__global__ void mlp3_kernel(const float* __restrict__ z2,
                            const float* __restrict__ W3,
                            const float* __restrict__ b3,
                            const float* __restrict__ true_vals,
                            float* __restrict__ out) {
    int tid = threadIdx.x;
    if (tid < BB) {
        float acc = b3[0];
        const float* zv = z2 + (size_t)tid * (2 * HH);
        for (int k = 0; k < 2 * HH; ++k) acc = fmaf(zv[k], W3[k], acc);
        out[tid] = 1.0f / (1.0f + __expf(-acc));        // predict [B,1]
    } else if (tid < 2 * BB) {
        out[tid] = true_vals[tid - BB];                 // true_vals [B]
    }
}

// ---------------------------------------------------------------------------
// Host launcher
// ---------------------------------------------------------------------------
extern "C" void kernel_launch(void* const* d_in, const int* in_sizes, int n_in,
                              void* d_out, int out_size, void* d_ws, size_t ws_size,
                              hipStream_t stream) {
    const float* x         = (const float*)d_in[0];
    const float* true_vals = (const float*)d_in[1];
    const float* W_ih      = (const float*)d_in[2];
    const float* W_hh      = (const float*)d_in[3];
    const float* b_ih      = (const float*)d_in[4];
    const float* b_hh      = (const float*)d_in[5];
    const float* W1        = (const float*)d_in[6];
    const float* b1        = (const float*)d_in[7];
    const float* W2        = (const float*)d_in[8];
    const float* b2        = (const float*)d_in[9];
    const float* W3        = (const float*)d_in[10];
    const float* b3        = (const float*)d_in[11];
    const int*   starts    = (const int*)d_in[12];
    const int*   ends      = (const int*)d_in[13];
    float* out = (float*)d_out;

    // Workspace layout (bytes). Total ~144.3 MB.
    char* ws = (char*)d_ws;
    const size_t off_xb   = 0;                                        // bf16 x
    const size_t off_wib  = off_xb  + (size_t)BB * TT * DD * 2;       // bf16 W_ih
    const size_t off_xw   = off_wib + (size_t)G4 * DD * 2;            // fp32 XW
    const size_t off_hbuf = off_xw  + (size_t)BB * TT * G4 * 4;       // fp32 h
    const size_t off_bar  = off_hbuf + (size_t)HH * 4;                // barrier
    const size_t off_hall = off_bar  + 256;                           // fp32 h_all
    const size_t off_z1   = off_hall + (size_t)BB * HH * 4;
    const size_t off_z2   = off_z1   + (size_t)BB * 2 * HH * 4;

    __bf16* xb   = (__bf16*)(ws + off_xb);
    __bf16* wib  = (__bf16*)(ws + off_wib);
    float*  XW   = (float*)(ws + off_xw);
    float*  hbuf = (float*)(ws + off_hbuf);
    int*    bar  = (int*)(ws + off_bar);
    float*  hall = (float*)(ws + off_hall);
    float*  z1   = (float*)(ws + off_z1);
    float*  z2   = (float*)(ws + off_z2);

    // 1) bf16 conversions + barrier init
    {
        long long total = (long long)BB * TT * DD + (long long)G4 * DD;
        int blocks = (int)((total + 255) / 256);
        convert_init_kernel<<<blocks, 256, 0, stream>>>(x, W_ih, xb, wib, bar);
    }
    // 2) WMMA GEMM: XW = x @ W_ih^T + b_ih + b_hh  (M=16384, N=2048, K=256)
    {
        dim3 grid(BB * TT / 128, G4 / 128);   // 128 x 16 blocks
        xw_gemm_kernel<<<grid, 256, 0, stream>>>(xb, wib, b_ih, b_hh, XW);
    }
    // 3) Sequential LSTM, W_hh sharded across 16 WGPs' LDS
    {
        size_t smem = (size_t)(512 * 128 + 512 + 256 + 128 + 32) * sizeof(float); // ~260 KB
        lstm_seq_kernel<<<16, 256, smem, stream>>>(XW, W_hh, starts, ends, hbuf, bar, hall);
    }
    // 4) MLP head
    mlp1_kernel<<<(BB * 2 * HH + 255) / 256, 256, 0, stream>>>(hall, W1, b1, z1);
    mlp2_kernel<<<(BB * 2 * HH + 255) / 256, 256, 0, stream>>>(z1, W2, b2, z2);
    mlp3_kernel<<<1, 256, 0, stream>>>(z2, W3, b3, true_vals, out);
}
